// PFAModel_661424963845
// MI455X (gfx1250) — matrix operand order repaired and checked by reading, not compiled, not run on hardware
//
#include <hip/hip_runtime.h>

// ---------------------------------------------------------------------------
// PFA forward on MI455X (gfx1250): linear-space forward algorithm as a chain
// of f16 WMMA GEMMs with f32 accumulation.
//   P = softmax(T_logits, axis=-1) is row-stochastic => alpha stays normalized
//   alpha_{t+1}[b,:] = alpha_t[b,:] @ P[sym_b]   (folded via augmented K=2048)
//   out[b] = clamp(sum_q alpha_T[b,q] * sigmoid(f[q]), 0, 1)
// Data movement: B slice + per-step alpha via Tensor Data Mover; x via
// global_load_async_to_lds; GEMM operands from LDS with software pipelining.
// ---------------------------------------------------------------------------

typedef _Float16 v16h __attribute__((ext_vector_type(16)));
typedef _Float16 v8h  __attribute__((ext_vector_type(8)));
typedef float    v8f  __attribute__((ext_vector_type(8)));
typedef unsigned int v4u __attribute__((ext_vector_type(4)));
typedef int      v8i  __attribute__((ext_vector_type(8)));
typedef int      v4i  __attribute__((ext_vector_type(4)));

#define PFA_B 256
#define PFA_T 512
#define PFA_Q 512
#define PFA_S 4
#define KAUG  (PFA_S * PFA_Q)        // 2048 augmented K
#define NCHUNK 64                     // output cols per workgroup
#define WGS_PER_GROUP 8               // 512 / 64
#define NGROUPS 16                    // 256 / 16 batch rows

// LDS carve (dynamic shared memory; offset 0 == wave LDS base)
#define B_LDS_OFF   0
#define B_LDS_BYTES (128 * NCHUNK * 16 * 2)    // 262144: [kb2][col][16] f16
#define A_LDS_OFF   (B_LDS_OFF + B_LDS_BYTES)
#define A_ROW_H     520                         // padded row stride (halves)
#define A_LDS_BYTES (16 * A_ROW_H * 2)          // 16640 (1024B data + 16B pad)
#define X_LDS_OFF   (A_LDS_OFF + A_LDS_BYTES)
#define X_LDS_BYTES (16 * PFA_T * 4)            // 32768 (raw int32 symbols)
#define LEN_OFF     (X_LDS_OFF + X_LDS_BYTES)
#define RED_OFF     (LEN_OFF + 64)
#define SYM_OFF     (RED_OFF + 512)
#define SMEM_TOTAL  (SYM_OFF + 32)              // 312160 < 320K

// Workspace layout (bytes)
#define WS_BPACK_OFF  0
#define WS_BPACK_SZ   ((size_t)KAUG * PFA_Q * 2)              // 2 MB
#define WS_ALPHA_OFF  (WS_BPACK_OFF + WS_BPACK_SZ)
#define WS_ALPHA_SZ   ((size_t)2 * PFA_B * PFA_Q * 2)         // 512 KB dbl buf
#define WS_ARRIVE_OFF (WS_ALPHA_OFF + WS_ALPHA_SZ)

// ---------------------------------------------------------------------------
// TDM: issue one 2D tensor_load_to_lds. D# per CDNA5 ISA.
// tile rows x cols (elements of 2B), global row stride in elements, optional
// LDS padding (pad codes: interval 7 -> 256 DW = 1024B, amount 3 -> 4 DW=16B).
// Toolchain here exposes the 6-arg builtin:
//   (uint32x4 g0, int32x8 g1, int32x4, int32x4, int32x8, i32 cpol)
// ---------------------------------------------------------------------------
__device__ __forceinline__ void tdm_load_2d(unsigned lds_off, const void* gaddr,
                                            unsigned tile_d0, unsigned tile_d1,
                                            unsigned tdim0, unsigned tdim1,
                                            unsigned stride0, bool pad_en,
                                            unsigned pad_int, unsigned pad_amt) {
  unsigned long long ga = (unsigned long long)gaddr;
  v4u g0;
  g0[0] = 1u;                                   // count=1, user descriptor
  g0[1] = lds_off;                              // LDS byte address
  g0[2] = (unsigned)(ga & 0xFFFFFFFFu);         // global_addr[31:0]
  g0[3] = (unsigned)((ga >> 32) & 0x01FFFFFFu)  // global_addr[56:32]
          | (2u << 30);                         // type=2 ("image")
  v8i g1;
  g1[0] = (int)((1u << 16)                      // data_size = 2 bytes
                | (pad_en ? (1u << 20) : 0u)
                | (pad_int << 22) | (pad_amt << 25));
  g1[1] = (int)((tdim0 & 0xFFFFu) << 16);       // tensor_dim0[15:0] @ bits63:48
  g1[2] = (int)(((tdim0 >> 16) & 0xFFFFu) | ((tdim1 & 0xFFFFu) << 16));
  g1[3] = (int)(((tdim1 >> 16) & 0xFFFFu) | ((tile_d0 & 0xFFFFu) << 16));
  g1[4] = (int)(tile_d1 & 0xFFFFu);             // tile_dim1; tile_dim2=0
  g1[5] = (int)stride0;                         // tensor_dim0_stride[31:0]
  g1[6] = 0;                                    // stride0[47:32], stride1 lo
  g1[7] = 0;
  v4i gz4 = {0, 0, 0, 0};
  v8i gz8 = {0, 0, 0, 0, 0, 0, 0, 0};
  __builtin_amdgcn_tensor_load_to_lds(g0, g1, gz4, gz4, gz8, 0);
}

// ---------------------------------------------------------------------------
// Init: zero arrival counters, alpha buffer 0 = one-hot(state 0). Runs every
// launch so graph replays self-reset.
// ---------------------------------------------------------------------------
__global__ void pfa_init(_Float16* __restrict__ alpha0, int* __restrict__ arrive) {
  int i = blockIdx.x * blockDim.x + threadIdx.x;
  int nthr = gridDim.x * blockDim.x;
  for (int j = i; j < NGROUPS * PFA_T; j += nthr) arrive[j] = 0;
  for (int j = i; j < PFA_B * PFA_Q; j += nthr)
    alpha0[j] = (_Float16)(((j & (PFA_Q - 1)) == 0) ? 1.0f : 0.0f);
}

// ---------------------------------------------------------------------------
// Pack: softmax each (q', s) row of T_logits over q, emit f16 in WMMA-B
// fragment order with a 16B half-slot XOR swizzle on column bit 3:
//   half index = ((k>>4)*Q + q)*16 + ((k&15) ^ (((q>>3)&1)<<3)),  k = s*Q+q'.
// ---------------------------------------------------------------------------
__global__ void pfa_pack(const float* __restrict__ Tl, _Float16* __restrict__ Bpack) {
  __shared__ float red[64];
  const int r = blockIdx.x;            // r = q'*4 + s   ([Q][S][Q] row-major)
  const int tid = threadIdx.x;
  const float* row = Tl + (size_t)r * PFA_Q;

  float mx = -3.4e38f;
  for (int q = tid; q < PFA_Q; q += 64) mx = fmaxf(mx, row[q]);
  red[tid] = mx; __syncthreads();
  for (int off = 32; off; off >>= 1) {
    if (tid < off) red[tid] = fmaxf(red[tid], red[tid + off]);
    __syncthreads();
  }
  mx = red[0]; __syncthreads();

  float sm = 0.f;
  for (int q = tid; q < PFA_Q; q += 64) sm += __expf(row[q] - mx);
  red[tid] = sm; __syncthreads();
  for (int off = 32; off; off >>= 1) {
    if (tid < off) red[tid] += red[tid + off];
    __syncthreads();
  }
  const float inv = 1.0f / red[0];

  const int s = r & 3, qp = r >> 2;
  const int k = s * PFA_Q + qp;
  _Float16* dstb = Bpack + ((size_t)(k >> 4) * PFA_Q) * 16;
  const int e = k & 15;
  for (int q = tid; q < PFA_Q; q += 64)
    dstb[(size_t)q * 16 + (e ^ (((q >> 3) & 1) << 3))] =
        (_Float16)(__expf(row[q] - mx) * inv);
}

// ---------------------------------------------------------------------------
// Main persistent kernel: 128 WGs x 128 threads (4 waves).
//   WG g: batch block grp=g>>3 (16 rows), column chunk nch=g&7 (64 cols).
//   Wave w owns the 16x16 tile at cols [nch*64 + 16w, +16), K = 2048.
// ---------------------------------------------------------------------------
__global__ void __launch_bounds__(128, 1)
pfa_forward(const int* __restrict__ x, const int* __restrict__ lengths,
            const float* __restrict__ f_logits,
            const _Float16* __restrict__ Bpack,
            _Float16* __restrict__ alphaBuf,
            int* __restrict__ arrive, float* __restrict__ out) {
  extern __shared__ char smem[];
  _Float16* B_lds  = (_Float16*)(smem + B_LDS_OFF);
  _Float16* a_lds  = (_Float16*)(smem + A_LDS_OFF);
  int*   x_lds  = (int*)(smem + X_LDS_OFF);
  int*   len_sh = (int*)(smem + LEN_OFF);
  float* red    = (float*)(smem + RED_OFF);
  unsigned char* symb = (unsigned char*)(smem + SYM_OFF);

  const int tid  = threadIdx.x;
  const int g    = blockIdx.x;
  const int grp  = g >> 3;
  const int nch  = g & 7;
  const int rb   = grp * 16;
  const int colbase = nch * NCHUNK;
  const int lane = tid & 31;
  const int wv   = tid >> 5;
  const int M    = lane & 15;              // A-row / D-col inside tile
  const int hi   = lane >> 4;
  const int c    = wv * 16 + M;            // local B column 0..63
  const int gsw  = (c >> 3) & 1;           // bank swizzle group

  // ---- one-time: x block (raw int32) via async-to-LDS ----
  for (int ch = tid; ch < 16 * 128; ch += 128) {   // 2048 chunks of 16 B
    int m = ch >> 7, j = ch & 127;
    const int* src = x + (size_t)(rb + m) * PFA_T + j * 4;
    unsigned lds_off = (unsigned)(X_LDS_OFF + (m * PFA_T + j * 4) * 4);
    asm volatile("global_load_async_to_lds_b128 %0, %1, off"
                 :: "v"(lds_off), "v"(src) : "memory");
  }
  // ---- one-time: B slice (2048 x 64 f16 = 256 KB) via one TDM op ----
  if (tid < 32) {
    tdm_load_2d(B_LDS_OFF, Bpack + (size_t)colbase * 16,
                /*tile*/ 1024, 128, /*tensor*/ 8192, 128, /*stride*/ 8192,
                false, 0, 0);
    __builtin_amdgcn_s_wait_tensorcnt(0);
  }
  if (tid < 16) len_sh[tid] = lengths[rb + tid];
  asm volatile("s_wait_asynccnt 0" ::: "memory");
  __syncthreads();

  int maxlen = 0;
  for (int m = 0; m < 16; ++m) maxlen = max(maxlen, len_sh[m]);
  if (maxlen > PFA_T) maxlen = PFA_T;

  v16h zero16;
#pragma unroll
  for (int e = 0; e < 16; ++e) zero16[e] = (_Float16)0.f;

  for (int t = 0; t < maxlen; ++t) {
    _Float16* cur = alphaBuf + (size_t)(t & 1) * (PFA_B * PFA_Q);
    _Float16* nxt = alphaBuf + (size_t)((t + 1) & 1) * (PFA_B * PFA_Q);

    // ---- stage alpha row-block (16x512 f16) into padded LDS rows via TDM ----
    if (tid < 32) {
      tdm_load_2d(A_LDS_OFF, cur + (size_t)rb * PFA_Q,
                  /*tile*/ PFA_Q, 16, /*tensor*/ PFA_Q, 16, /*stride*/ PFA_Q,
                  /*pad*/ true, /*interval 1024B*/ 7, /*amount 16B*/ 3);
      __builtin_amdgcn_s_wait_tensorcnt(0);
    }
    if (tid < 16) symb[tid] = (unsigned char)x_lds[tid * PFA_T + t];
    __syncthreads();

    unsigned present = 0;
    for (int m = 0; m < 16; ++m) present |= 1u << symb[m];

    v8f acc = {};
    const int mysym = (int)symb[M];
    const _Float16* arow = a_lds + M * A_ROW_H + 16 * hi;

    for (int s = 0; s < PFA_S; ++s) {
      if (!((present >> s) & 1u)) continue;    // uniform skip, EXEC stays full
      const bool on = (mysym == s);            // per-lane gating via zero A rows
      const _Float16* Bs = B_lds + ((size_t)(s * 32 + hi) * NCHUNK + c) * 16;

      // software pipeline: prefetch next iteration's fragments during WMMA
      v8h pa0 = *(const v8h*)(arow + 0);
      v8h pa1 = *(const v8h*)(arow + 8);
      v8h pb0 = *(const v8h*)(Bs + 8 * gsw);
      v8h pb1 = *(const v8h*)(Bs + 8 - 8 * gsw);
#pragma unroll
      for (int kb = 0; kb < PFA_Q; kb += 32) {
        const int kbn = (kb + 32) & (PFA_Q - 1);       // wraps on last iter
        v8h na0 = *(const v8h*)(arow + kbn);
        v8h na1 = *(const v8h*)(arow + kbn + 8);
        const _Float16* Bn = Bs + (size_t)(kbn >> 4) * (NCHUNK * 16);
        v8h nb0 = *(const v8h*)(Bn + 8 * gsw);
        v8h nb1 = *(const v8h*)(Bn + 8 - 8 * gsw);

        v16h a = on ? __builtin_shufflevector(pa0, pa1, 0, 1, 2, 3, 4, 5, 6, 7,
                                              8, 9, 10, 11, 12, 13, 14, 15)
                    : zero16;
        v16h b = __builtin_shufflevector(pb0, pb1, 0, 1, 2, 3, 4, 5, 6, 7,
                                         8, 9, 10, 11, 12, 13, 14, 15);
        acc = __builtin_amdgcn_wmma_f32_16x16x32_f16(
            false, a, false, b, (short)0, acc, false, false);
        pa0 = na0; pa1 = na1; pb0 = nb0; pb1 = nb1;
      }
    }

    // ---- masked writeback: D[v] = row (v + 8*hi), col (lane&15) ----
    const int gcol = colbase + c;
#pragma unroll
    for (int v = 0; v < 8; ++v) {
      const int m = v + 8 * hi;
      _Float16 val = (t < len_sh[m]) ? (_Float16)acc[v]
                                     : a_lds[m * A_ROW_H + gcol];
      nxt[(size_t)(rb + m) * PFA_Q + gcol] = val;
    }

    // ---- sync the 8 WGs of this batch block ----
    __threadfence();
    __builtin_amdgcn_s_cluster_barrier();      // NOP unless cluster-dispatched
    if (tid == 0) {
      int* a = arrive + grp * PFA_T + t;
      __hip_atomic_fetch_add(a, 1, __ATOMIC_RELEASE, __HIP_MEMORY_SCOPE_AGENT);
      while (__hip_atomic_load(a, __ATOMIC_ACQUIRE, __HIP_MEMORY_SCOPE_AGENT) <
             WGS_PER_GROUP)
        __builtin_amdgcn_s_sleep(1);
    }
    __syncthreads();
  }

  // ---- finalize: out[b] = clamp(sum_q alpha * sigmoid(f), 0, 1) ----
  if (nch == 0) {
    const _Float16* af = alphaBuf + (size_t)(maxlen & 1) * (PFA_B * PFA_Q);
    const int m = tid >> 3, c0 = tid & 7;
    float part = 0.f;
    for (int q = c0; q < PFA_Q; q += 8) {
      float fa = (float)af[(size_t)(rb + m) * PFA_Q + q];
      float fl = f_logits[q];
      part += fa * (1.0f / (1.0f + __expf(-fl)));
    }
    red[tid] = part;
    __syncthreads();
    if ((tid & 7) == 0) {
      float s2 = 0.f;
      for (int j = 0; j < 8; ++j) s2 += red[tid + j];  // deterministic tree
      out[rb + m] = fminf(fmaxf(s2, 0.f), 1.f);
    }
  }
}

// ---------------------------------------------------------------------------
extern "C" void kernel_launch(void* const* d_in, const int* in_sizes, int n_in,
                              void* d_out, int out_size, void* d_ws, size_t ws_size,
                              hipStream_t stream) {
  const int*   x        = (const int*)d_in[0];     // [256,512]
  const int*   lengths  = (const int*)d_in[1];     // [256]
  const float* T_logits = (const float*)d_in[2];   // [512,4,512]
  const float* f_logits = (const float*)d_in[3];   // [512]
  float* out = (float*)d_out;

  char* ws = (char*)d_ws;
  _Float16* Bpack  = (_Float16*)(ws + WS_BPACK_OFF);
  _Float16* alphaB = (_Float16*)(ws + WS_ALPHA_OFF);
  int*      arrive = (int*)(ws + WS_ARRIVE_OFF);

  pfa_init<<<32, 256, 0, stream>>>(alphaB, arrive);
  pfa_pack<<<PFA_Q * PFA_S, 64, 0, stream>>>(T_logits, Bpack);
  pfa_forward<<<NGROUPS * WGS_PER_GROUP, 128, SMEM_TOTAL, stream>>>(
      x, lengths, f_logits, Bpack, alphaB, arrive, out);
}
